// HungarianMatcher_65850438582451
// MI455X (gfx1250) — matrix operand order repaired and checked by reading, not compiled, not run on hardware
//
#include <hip/hip_runtime.h>
#include <hip/hip_bf16.h>
#include <cstdint>
#include <cstddef>

// ---- problem constants (match reference) ----
#define F_ALPHA 0.25f
#define F_EPS   1e-8f
constexpr int B_ = 64, Q_ = 900, T_ = 300, C_ = 80;
constexpr int QP = 912;   // 57 * 16  (Q padded to tile)
constexpr int CP = 96;    // 3 * 32   (C padded to WMMA K)
constexpr int TP = 304;   // 19 * 16  (T padded to tile)
constexpr int QT = QP / 16;   // 57 q-tiles
constexpr int TT = TP / 16;   // 19 t-tiles

typedef __attribute__((ext_vector_type(16))) __bf16 v16bf;
typedef __attribute__((ext_vector_type(8)))  float  v8f;

union ABu { v16bf bf; unsigned int u[8]; };

// ---------------------------------------------------------------------------
// Kernel 1a: focal class-cost table ccost[b, qpad, cpad] (bf16, W_CLASS folded)
// ---------------------------------------------------------------------------
__global__ void __launch_bounds__(256)
focal_table_kernel(const float* __restrict__ logits,
                   unsigned short* __restrict__ ccost) {
  long e = (long)blockIdx.x * blockDim.x + threadIdx.x;
  const long total = (long)B_ * QP * CP;
  if (e >= total) return;
  int c = (int)(e % CP);
  int q = (int)((e / CP) % QP);
  int b = (int)(e / ((long)CP * QP));
  float val = 0.0f;
  if (c < C_ && q < Q_) {
    float x   = logits[((long)b * Q_ + q) * C_ + c];
    float p   = 1.0f / (1.0f + __expf(-x));
    float omp = 1.0f - p;
    float pos = F_ALPHA * omp * omp * (-__logf(p + F_EPS));
    float neg = (1.0f - F_ALPHA) * p * p * (-__logf(omp + F_EPS));
    val = 2.0f * (pos - neg);              // W_CLASS = 2 folded in
  }
  unsigned u = __float_as_uint(val);
  unsigned r = (u + 0x7FFFu + ((u >> 16) & 1u)) >> 16;   // RNE to bf16
  ccost[e] = (unsigned short)r;
}

// ---------------------------------------------------------------------------
// Kernel 1b: precompute one-hot B operands in WMMA B-register layout.
// bmat[((b*TT + tt)*3 + kblk)*256 + lane*8 + v] -- lane reads 32B contiguous.
// 16-bit B 32x16 layout, contiguous k (as load_a): lane l: N=l&15, h=l>>4; VGPR v holds
// (K=16h+2v, K=16h+2v+1) packed lo/hi.  Total: 64*19*3*256 dwords (~3.7 MB).
// ---------------------------------------------------------------------------
__global__ void __launch_bounds__(256)
onehot_kernel(const int* __restrict__ tgt_labels,
              unsigned int* __restrict__ bmat) {
  int e = blockIdx.x * 256 + threadIdx.x;      // one dword each
  const int total = B_ * TT * 3 * 256;
  if (e >= total) return;
  int v    = e & 7;
  int lane = (e >> 3) & 31;
  int g    = e >> 8;                            // (b*TT + tt)*3 + kblk
  int kblk = g % 3;
  int tt   = (g / 3) % TT;
  int b    = g / (3 * TT);

  int h = lane >> 4;
  int n = lane & 15;
  int t = tt * 16 + n;
  int lab = (t < T_) ? tgt_labels[(size_t)b * T_ + t] : -1;
  int rel = lab - kblk * 32 - 16 * h;
  unsigned urel = (unsigned)rel;
  unsigned word = 0u;
  if (urel < 16u && (int)(urel >> 1) == v)
    word = (urel & 1u) ? 0x3F800000u : 0x00003F80u;   // bf16 1.0 in hi/lo half
  bmat[e] = word;
}

// ---------------------------------------------------------------------------
// helpers for kernel 2
// ---------------------------------------------------------------------------
// A operand (16x32 bf16), contiguous-k A layout (as bmat):
// lane l: M = l&15, h = l>>4; v0..3 <- K = 16h..16h+7 ; v4..7 <- K = 16h+8..+7
__device__ __forceinline__ ABu load_a(const unsigned short* __restrict__ arow,
                                      int kb, int h) {
  ABu d;
  uint4 lo = *(const uint4*)(arow + kb + 16 * h);
  uint4 hi = *(const uint4*)(arow + kb + 16 * h + 8);
  d.u[0] = lo.x; d.u[1] = lo.y; d.u[2] = lo.z; d.u[3] = lo.w;
  d.u[4] = hi.x; d.u[5] = hi.y; d.u[6] = hi.z; d.u[7] = hi.w;
  return d;
}

// B operand: two coalesced b128 loads from the precomputed table
__device__ __forceinline__ ABu load_b(const unsigned int* __restrict__ p) {
  ABu d;
  uint4 lo = *(const uint4*)(p);
  uint4 hi = *(const uint4*)(p + 4);
  d.u[0] = lo.x; d.u[1] = lo.y; d.u[2] = lo.z; d.u[3] = lo.w;
  d.u[4] = hi.x; d.u[5] = hi.y; d.u[6] = hi.z; d.u[7] = hi.w;
  return d;
}

// ---------------------------------------------------------------------------
// Kernel 2: fused bbox-L1 + GIoU (VALU, in C/D layout) + class cost (WMMA)
// grid = B_ * QT blocks of 256 threads (8 waves).
// Each wave handles t-tiles {wave, wave+8, wave+16(<TT)} -- fully unrolled,
// with ALL B-operand loads issued up-front so no wmma waits on a fresh load.
// ---------------------------------------------------------------------------
__global__ void __launch_bounds__(256)
cost_kernel(const float* __restrict__ pred_boxes,
            const float* __restrict__ tgt_boxes,
            const unsigned short* __restrict__ ccost,
            const unsigned int* __restrict__ bmat,
            float* __restrict__ out) {
  __shared__ float4 sRaw[TP];                       // async-staged tgt boxes
  // target SoA: xyxy, sums (2cx,2cy), w, h
  __shared__ float sTx0[TP], sTy0[TP], sTx1[TP], sTy1[TP];
  __shared__ float sTsx[TP], sTsy[TP], sTw[TP], sTh[TP];
  __shared__ float sP[4][16];                       // pred xyxy (16 q rows)

  const int tid  = threadIdx.x;
  const int b    = blockIdx.x / QT;
  const int q0   = (blockIdx.x % QT) * 16;
  const int lane = tid & 31;
  const int h    = lane >> 4;
  const int m    = lane & 15;
  const int wave = tid >> 5;

  // --- stage tgt boxes for this batch into LDS via CDNA5 async copy ---
  {
    unsigned long long gbase =
        (unsigned long long)(tgt_boxes + (size_t)b * T_ * 4);
    for (int t = tid; t < T_; t += 256) {
      unsigned ldsOff = (unsigned)(size_t)(&sRaw[t]);
      unsigned voff   = (unsigned)(t * 16);
      asm volatile("global_load_async_to_lds_b128 %0, %1, %2"
                   :: "v"(ldsOff), "v"(voff), "s"(gbase) : "memory");
    }
    __builtin_amdgcn_s_wait_asynccnt(0);
  }
  __syncthreads();

  // --- derive SoA target-box features (pad region zero-filled) ---
  for (int t = tid; t < TP; t += 256) {
    float cx = 0.f, cy = 0.f, w = 0.f, hh = 0.f;
    if (t < T_) {
      float4 r = sRaw[t];
      cx = r.x; cy = r.y; w = r.z; hh = r.w;
    }
    sTx0[t] = cx - 0.5f * w;  sTy0[t] = cy - 0.5f * hh;
    sTx1[t] = cx + 0.5f * w;  sTy1[t] = cy + 0.5f * hh;
    sTsx[t] = 2.0f * cx;      sTsy[t] = 2.0f * cy;
    sTw[t]  = w;              sTh[t]  = hh;
  }
  if (tid < 16) {
    int q = q0 + tid; if (q >= Q_) q = Q_ - 1;      // dup rows (masked at store)
    const float* pb = pred_boxes + ((size_t)b * Q_ + q) * 4;
    float cx = pb[0], cy = pb[1], w = pb[2], hh = pb[3];
    sP[0][tid] = cx - 0.5f * w;  sP[1][tid] = cy - 0.5f * hh;
    sP[2][tid] = cx + 0.5f * w;  sP[3][tid] = cy + 0.5f * hh;
  }
  __syncthreads();

  // --- A operand (class-cost tile, constant across t-tiles) ---
  const unsigned short* arow = ccost + ((size_t)b * QP + (q0 + m)) * CP;
  ABu a0 = load_a(arow, 0, h);
  ABu a1 = load_a(arow, 32, h);
  ABu a2 = load_a(arow, 64, h);

  // --- pred-box xyxy for this lane's 8 output rows (M = j + 8h) ---
  float px0[8], py0[8], px1[8], py1[8];
#pragma unroll
  for (int j = 0; j < 8; ++j) {
    int mm = j + 8 * h;
    px0[j] = sP[0][mm]; py0[j] = sP[1][mm];
    px1[j] = sP[2][mm]; py1[j] = sP[3][mm];
  }

  // --- issue ALL B loads for this wave's tiles up-front (L2-resident) ---
  const int tt0 = wave, tt1 = wave + 8, tt2 = wave + 16;
  const bool has2 = (tt2 < TT);                     // uniform per wave
  const unsigned int* bbase = bmat + ((size_t)b * TT * 3) * 256 + lane * 8;
  const unsigned int* bb0 = bbase + (size_t)tt0 * 3 * 256;
  const unsigned int* bb1 = bbase + (size_t)tt1 * 3 * 256;
  ABu c00 = load_b(bb0), c01 = load_b(bb0 + 256), c02 = load_b(bb0 + 512);
  ABu c10 = load_b(bb1), c11 = load_b(bb1 + 256), c12 = load_b(bb1 + 512);
  ABu c20 = {}, c21 = {}, c22 = {};
  if (has2) {
    const unsigned int* bb2 = bbase + (size_t)tt2 * 3 * 256;
    c20 = load_b(bb2); c21 = load_b(bb2 + 256); c22 = load_b(bb2 + 512);
  }

  auto body = [&](int tt, const ABu& B0, const ABu& B1, const ABu& B2) {
    const int t = tt * 16 + m;                      // this lane's column (N = m)
    float tx0 = sTx0[t], ty0 = sTy0[t], tx1 = sTx1[t], ty1 = sTy1[t];
    float tsx = sTsx[t], tsy = sTsy[t], tw = sTw[t], th_ = sTh[t];
    float ta = tw * th_;

    v8f acc;
#pragma unroll
    for (int j = 0; j < 8; ++j) {
      float pwd = px1[j] - px0[j], phd = py1[j] - py0[j];
      float pa  = pwd * phd;
      // GIoU
      float ltx = fmaxf(px0[j], tx0), lty = fmaxf(py0[j], ty0);
      float rbx = fminf(px1[j], tx1), rby = fminf(py1[j], ty1);
      float iw = fmaxf(rbx - ltx, 0.0f), ih = fmaxf(rby - lty, 0.0f);
      float inter = iw * ih;
      float uni = pa + ta - inter;
      float iou = inter * __builtin_amdgcn_rcpf(uni);
      float cx0 = fminf(px0[j], tx0), cy0 = fminf(py0[j], ty0);
      float cx1 = fmaxf(px1[j], tx1), cy1 = fmaxf(py1[j], ty1);
      float cw = fmaxf(cx1 - cx0, 0.0f), chh = fmaxf(cy1 - cy0, 0.0f);
      float ac = cw * chh;
      float giou = iou - (ac - uni) * __builtin_amdgcn_rcpf(ac);
      // L1 on cxcywh from xyxy: |dcx| = 0.5|sum_x diff|, w = x1-x0
      float l1 = 0.5f * (fabsf(px0[j] + px1[j] - tsx) +
                         fabsf(py0[j] + py1[j] - tsy)) +
                 fabsf(pwd - tw) + fabsf(phd - th_);
      acc[j] = 5.0f * l1 - 2.0f * giou;             // W_BBOX, W_GIOU
    }

    // class cost via one-hot bf16 WMMA, fused into the f32 accumulator
    acc = __builtin_amdgcn_wmma_f32_16x16x32_bf16(false, a0.bf, false, B0.bf,
                                                  (short)0, acc, false, false);
    acc = __builtin_amdgcn_wmma_f32_16x16x32_bf16(false, a1.bf, false, B1.bf,
                                                  (short)0, acc, false, false);
    acc = __builtin_amdgcn_wmma_f32_16x16x32_bf16(false, a2.bf, false, B2.bf,
                                                  (short)0, acc, false, false);

    // nan_to_num (nan->1, +/-inf -> +/-1e6) + store
    const bool fast = (tt * 16 + 15 < T_) && (q0 + 15 < Q_);  // uniform
    if (fast) {
#pragma unroll
      for (int j = 0; j < 8; ++j) {
        float cst = acc[j];
        cst = (cst == cst) ? fminf(fmaxf(cst, -1e6f), 1e6f) : 1.0f;
        out[((size_t)b * Q_ + q0 + j + 8 * h) * T_ + t] = cst;
      }
    } else if (t < T_) {
#pragma unroll
      for (int j = 0; j < 8; ++j) {
        int q = q0 + j + 8 * h;
        if (q < Q_) {
          float cst = acc[j];
          cst = (cst == cst) ? fminf(fmaxf(cst, -1e6f), 1e6f) : 1.0f;
          out[((size_t)b * Q_ + q) * T_ + t] = cst;
        }
      }
    }
  };

  body(tt0, c00, c01, c02);
  body(tt1, c10, c11, c12);
  if (has2) body(tt2, c20, c21, c22);
}

// ---------------------------------------------------------------------------
extern "C" void kernel_launch(void* const* d_in, const int* in_sizes, int n_in,
                              void* d_out, int out_size, void* d_ws, size_t ws_size,
                              hipStream_t stream) {
  (void)in_sizes; (void)n_in; (void)out_size; (void)ws_size;
  const float* pred_logits = (const float*)d_in[0];
  const float* pred_boxes  = (const float*)d_in[1];
  const int*   tgt_labels  = (const int*)d_in[2];   // integer input -> int32 per harness
  const float* tgt_boxes   = (const float*)d_in[3];
  float* out = (float*)d_out;

  // workspace layout: [ccost bf16 table | precomputed one-hot B operands]
  unsigned short* ccost = (unsigned short*)d_ws;            // B_*QP*CP*2 bytes
  const size_t ccost_bytes = (size_t)B_ * QP * CP * 2;      // 11,206,656 (16B aligned)
  unsigned int* bmat = (unsigned int*)((char*)d_ws + ccost_bytes); // ~3.7 MB

  const long total1 = (long)B_ * QP * CP;
  focal_table_kernel<<<(int)((total1 + 255) / 256), 256, 0, stream>>>(pred_logits, ccost);

  const int total2 = B_ * TT * 3 * 256;
  onehot_kernel<<<(total2 + 255) / 256, 256, 0, stream>>>(tgt_labels, bmat);

  cost_kernel<<<B_ * QT, 256, 0, stream>>>(pred_boxes, tgt_boxes,
                                           ccost, bmat, out);
}
